// E59bGatedHighwayCell_2147483648041
// MI455X (gfx1250) — matrix operand order repaired and checked by reading, not compiled, not run on hardware
//
#include <hip/hip_runtime.h>
#include <hip/hip_bf16.h>

// ---------------------------------------------------------------------------
// E59b GatedHighwayCell for MI455X (gfx1250, wave32, WMMA).
//   Phase 1: convert x, W, W_g to bf16 (workspace)
//   Phase 2: WMMA bf16 GEMM -> Wx_all (f32, +bias) and gate_all (f32, sigmoid)
//            block tile 128x256, 8 waves (2x4), 64x64 per wave (16 WMMA tiles)
//   Phase 3: sequential scan over T (one workgroup per batch b), RMSNorm+silu
//            with register double-buffering + global_prefetch of t+8
// ---------------------------------------------------------------------------

typedef __attribute__((ext_vector_type(16))) __bf16 v16bf;
typedef __attribute__((ext_vector_type(8)))  __bf16 v8bf;
typedef __attribute__((ext_vector_type(4)))  __bf16 v4bf;
typedef __attribute__((ext_vector_type(8)))  float  v8f;

#define TDIM 2048
#define BDIM 16
#define DDIM 1024
#define MDIM (TDIM * BDIM)   // 32768

__device__ __forceinline__ float fast_sigmoid(float x) {
    // v_exp_f32 + v_rcp_f32 (avoid IEEE div_scale/div_fmas expansion)
    return __builtin_amdgcn_rcpf(1.0f + __expf(-x));
}

// ---------------------------------------------------------------------------
// f32 -> bf16 conversion (4 elements / thread, vectorized)
// ---------------------------------------------------------------------------
__global__ __launch_bounds__(256) void f32_to_bf16_kernel(const float* __restrict__ in,
                                                          __bf16* __restrict__ out,
                                                          unsigned long long n) {
    unsigned long long i = ((unsigned long long)blockIdx.x * blockDim.x + threadIdx.x) * 4ull;
    if (i < n) {
        float4 v = *(const float4*)(in + i);
        v4bf o;
        o.x = (__bf16)v.x; o.y = (__bf16)v.y; o.z = (__bf16)v.z; o.w = (__bf16)v.w;
        *(v4bf*)(out + i) = o;
    }
}

// ---------------------------------------------------------------------------
// WMMA GEMM: out[m][n] = sum_k X[m][k] * Wmat[n][k]  (+ bias[n], opt. sigmoid)
// X:    [M, K] bf16 row-major (K contiguous)
// Wmat: [N, K] bf16 row-major (K contiguous)  -> B matrix K x N column-major
// Block tile 128x256, 256 threads = 8 waves in 2x4; each wave 64x64 (4x4 tiles)
// blockIdx.z: 0 -> Wx (bias b), 1 -> gate (bias b_g, sigmoid)
// ---------------------------------------------------------------------------
__global__ __launch_bounds__(256) void gemm_bf16_wmma_kernel(
        const __bf16* __restrict__ X,
        const __bf16* __restrict__ Wb,
        const __bf16* __restrict__ Wgb,
        const float* __restrict__ bias0,
        const float* __restrict__ bias1,
        float* __restrict__ wx_out,
        float* __restrict__ gate_out) {
    const int K = DDIM;
    const int N = DDIM;

    const bool  is_gate = (blockIdx.z != 0);
    const __bf16* __restrict__ Bmat = is_gate ? Wgb : Wb;
    const float*  __restrict__ bvec = is_gate ? bias1 : bias0;
    float* __restrict__ outp        = is_gate ? gate_out : wx_out;

    const int tid  = threadIdx.x;
    const int wave = tid >> 5;
    const int lane = tid & 31;
    const int half = lane >> 4;   // 0: lanes 0-15, 1: lanes 16-31
    const int r    = lane & 15;

    const int wm = wave >> 2;     // 0..1
    const int wn = wave & 3;      // 0..3

    const int m0 = blockIdx.y * 128 + wm * 64;  // wave M base (4 tiles of 16)
    const int n0 = blockIdx.x * 256 + wn * 64;  // wave N base (4 tiles of 16)

    v8f acc[4][4];
#pragma unroll
    for (int i = 0; i < 4; ++i)
#pragma unroll
        for (int j = 0; j < 4; ++j)
            acc[i][j] = (v8f)0.0f;

    for (int k0 = 0; k0 < K; k0 += 32) {
        // --- A fragments: 16x32 bf16. Lane holds row (m0+i*16+r).
        //     elems 0..7  = K = k0 + half*8 + {0..7}
        //     elems 8..15 = K = k0 + 16 + half*8 + {0..7}
        v16bf afrag[4];
#pragma unroll
        for (int i = 0; i < 4; ++i) {
            const __bf16* p = X + (unsigned long long)(m0 + i * 16 + r) * K + k0 + half * 8;
            v8bf lo = *(const v8bf*)p;          // K = k0 + half*8 .. +7
            v8bf hi = *(const v8bf*)(p + 16);   // K = k0 + 16 + half*8 .. +7
            afrag[i] = __builtin_shufflevector(lo, hi,
                0, 1, 2, 3, 4, 5, 6, 7, 8, 9, 10, 11, 12, 13, 14, 15);
        }
        // --- B fragments: 32x16 (KxN). Lane holds column (n0+j*16+r),
        //     elems 0..7 = K = k0 + half*8 + {0..7}, elems 8..15 = K = k0 + 16 + half*8 + {0..7}
        v16bf bfrag[4];
#pragma unroll
        for (int j = 0; j < 4; ++j) {
            const __bf16* p = Bmat + (unsigned long long)(n0 + j * 16 + r) * K + k0 + half * 8;
            v8bf lo = *(const v8bf*)p;
            v8bf hi = *(const v8bf*)(p + 16);
            bfrag[j] = __builtin_shufflevector(lo, hi,
                0, 1, 2, 3, 4, 5, 6, 7, 8, 9, 10, 11, 12, 13, 14, 15);
        }
#pragma unroll
        for (int i = 0; i < 4; ++i)
#pragma unroll
            for (int j = 0; j < 4; ++j)
                acc[i][j] = __builtin_amdgcn_wmma_f32_16x16x32_bf16(
                    /*neg_a=*/false, afrag[i],
                    /*neg_b=*/false, bfrag[j],
                    /*c_mod=*/(short)0, acc[i][j],
                    /*reuse_a=*/false, /*reuse_b=*/false);
    }

    // Epilogue. C layout: VGPR v -> M = v + 8*half, N = r (per 16x16 tile).
#pragma unroll
    for (int i = 0; i < 4; ++i) {
#pragma unroll
        for (int j = 0; j < 4; ++j) {
            const int n = n0 + j * 16 + r;
            const float bv = bvec[n];
#pragma unroll
            for (int v = 0; v < 8; ++v) {
                const int m = m0 + i * 16 + v + 8 * half;
                float val = acc[i][j][v] + bv;
                if (is_gate) val = fast_sigmoid(val);
                outp[(unsigned long long)m * N + n] = val;
            }
        }
    }
}

// ---------------------------------------------------------------------------
// Sequential scan: one workgroup per batch row b. 256 threads x 4 elems = D.
// h kept in registers; RMS reduction via shfl_xor (wave32) + LDS across waves.
// Next-step loads double-buffered in registers; t+8 prefetched into caches.
// ---------------------------------------------------------------------------
__global__ __launch_bounds__(256) void scan_kernel(
        const float* __restrict__ wx_all,    // [T, B, D]  (== [M, D])
        const float* __restrict__ gate_all,  // [T, B, D]
        const float* __restrict__ h0,        // [B, D]
        float* __restrict__ outs,            // [T, B, D]
        float* __restrict__ hout) {          // [T+1, B, D]
    const int b    = blockIdx.x;
    const int tid  = threadIdx.x;
    const int lane = tid & 31;
    const int wave = tid >> 5;
    const int d0   = tid * 4;

    __shared__ float red[8];

    const unsigned long long stride = (unsigned long long)BDIM * DDIM;
    const unsigned long long bd     = (unsigned long long)b * DDIM + d0;

    float4 h = *(const float4*)(h0 + bd);
    *(float4*)(hout + bd) = h;   // h[0] = h0

    unsigned long long base = bd;                 // (t*B + b)*D + d0 at t=0
    float4 g = *(const float4*)(gate_all + base);
    float4 w = *(const float4*)(wx_all + base);

    for (int t = 0; t < TDIM; ++t) {
        // Prefetch t+8 (one lane per 128B cacheline) -> global_prefetch_b8
        if ((t + 8 < TDIM) && ((tid & 7) == 0)) {
            __builtin_prefetch(gate_all + base + 8 * stride, 0, 1);
            __builtin_prefetch(wx_all + base + 8 * stride, 0, 1);
        }

        h.x += g.x * w.x;
        h.y += g.y * w.y;
        h.z += g.z * w.z;
        h.w += g.w * w.w;

        // Issue next step's loads early; they retire behind the reduction.
        float4 gn = g, wn = w;
        if (t + 1 < TDIM) {
            gn = *(const float4*)(gate_all + base + stride);
            wn = *(const float4*)(wx_all + base + stride);
        }

        float ss = h.x * h.x + h.y * h.y + h.z * h.z + h.w * h.w;
#pragma unroll
        for (int off = 16; off >= 1; off >>= 1)
            ss += __shfl_xor(ss, off, 32);
        if (lane == 0) red[wave] = ss;
        __syncthreads();
        float tot = red[0] + red[1] + red[2] + red[3] +
                    red[4] + red[5] + red[6] + red[7];
        const float rms = __builtin_amdgcn_rsqf(tot * (1.0f / (float)DDIM) + 1e-6f);

        h.x *= rms; h.y *= rms; h.z *= rms; h.w *= rms;

        // out = h * silu(h) = h^2 * sigmoid(h)
        float4 o;
        o.x = h.x * h.x * fast_sigmoid(h.x);
        o.y = h.y * h.y * fast_sigmoid(h.y);
        o.z = h.z * h.z * fast_sigmoid(h.z);
        o.w = h.w * h.w * fast_sigmoid(h.w);

        *(float4*)(hout + base + stride) = h;     // (t+1)*B*D + b*D + d0
        *(float4*)(outs + base) = o;

        base += stride;
        g = gn; w = wn;
        __syncthreads();  // protect red[] before next iteration
    }
}

// ---------------------------------------------------------------------------
// Launch
// ---------------------------------------------------------------------------
extern "C" void kernel_launch(void* const* d_in, const int* in_sizes, int n_in,
                              void* d_out, int out_size, void* d_ws, size_t ws_size,
                              hipStream_t stream) {
    (void)in_sizes; (void)n_in; (void)out_size; (void)ws_size;

    const float* x    = (const float*)d_in[0];  // [T, B, D]
    const float* h0   = (const float*)d_in[1];  // [B, D]
    const float* W    = (const float*)d_in[2];  // [D, D]
    const float* W_g  = (const float*)d_in[3];  // [D, D]
    const float* b    = (const float*)d_in[4];  // [D]
    const float* b_g  = (const float*)d_in[5];  // [D]

    char* ws = (char*)d_ws;
    const unsigned long long xbf_bytes = (unsigned long long)MDIM * DDIM * 2ull;
    const unsigned long long wbf_bytes = (unsigned long long)DDIM * DDIM * 2ull;

    __bf16* xbf   = (__bf16*)ws;
    __bf16* wbf   = (__bf16*)(ws + xbf_bytes);
    __bf16* wgbf  = (__bf16*)(ws + xbf_bytes + wbf_bytes);
    float*  wx_all   = (float*)(ws + xbf_bytes + 2ull * wbf_bytes);
    float*  gate_all = wx_all + (unsigned long long)MDIM * DDIM;

    // Phase 1: conversions
    {
        unsigned long long n = (unsigned long long)MDIM * DDIM;
        f32_to_bf16_kernel<<<(unsigned)((n / 4 + 255) / 256), 256, 0, stream>>>(x, xbf, n);
        unsigned long long nw = (unsigned long long)DDIM * DDIM;
        f32_to_bf16_kernel<<<(unsigned)((nw / 4 + 255) / 256), 256, 0, stream>>>(W, wbf, nw);
        f32_to_bf16_kernel<<<(unsigned)((nw / 4 + 255) / 256), 256, 0, stream>>>(W_g, wgbf, nw);
    }

    // Phase 2: both projections (z=0 -> Wx+b, z=1 -> sigmoid(Wg x + b_g))
    {
        dim3 grid(DDIM / 256, MDIM / 128, 2);
        gemm_bf16_wmma_kernel<<<grid, 256, 0, stream>>>(xbf, wbf, wgbf, b, b_g,
                                                        wx_all, gate_all);
    }

    // Phase 3: recurrence
    {
        float* outs = (float*)d_out;                                 // [T, B, D]
        float* hout = outs + (unsigned long long)TDIM * BDIM * DDIM; // [T+1, B, D]
        scan_kernel<<<BDIM, 256, 0, stream>>>(wx_all, gate_all, h0, outs, hout);
    }
}
